// DogeCDMoE_66872640799384
// MI455X (gfx1250) — compile-verified, matrix operand order checked by reading, not run on hardware
//
#include <hip/hip_runtime.h>
#include <hip/hip_bf16.h>

#define HID    1024
#define INTER  4096
#define NTOK   4096      // B*T = 2*2048
#define HEADS  4
#define RDIM   128
#define NKEYS  128
#define TOPK   8

typedef __attribute__((ext_vector_type(16))) __bf16 v16bf;
typedef __attribute__((ext_vector_type(4)))  __bf16 v4bf;
typedef __attribute__((ext_vector_type(8)))  float  v8f;

// ---------------------------------------------------------------------------
// CDNA5 async global->LDS copy (ASYNCcnt-tracked, no VGPR staging).
// VDST operand = 32-bit LDS byte offset; generic LDS pointers carry the offset
// in bits [31:0] (LDS aperture, ISA 10.2), so truncation yields the offset.
// ---------------------------------------------------------------------------
__device__ __forceinline__ void async_copy_b128(const unsigned int* gsrc,
                                                unsigned int* ldst) {
#if defined(__gfx1250__)
    unsigned loff = (unsigned)(unsigned long long)ldst;
    asm volatile("global_load_async_to_lds_b128 %0, %1, off"
                 :: "v"(loff), "v"(gsrc) : "memory");
#else
    ldst[0] = gsrc[0]; ldst[1] = gsrc[1]; ldst[2] = gsrc[2]; ldst[3] = gsrc[3];
#endif
}

// Async loads complete in issue order: with 8 copies per K-tile, waiting to 8
// retires exactly the previous tile while the next tile's copies stay in flight.
__device__ __forceinline__ void wait_async_le8() {
#if defined(__gfx1250__)
    asm volatile("s_wait_asynccnt 0x8" ::: "memory");
#endif
}
__device__ __forceinline__ void wait_async_le0() {
#if defined(__gfx1250__)
    asm volatile("s_wait_asynccnt 0x0" ::: "memory");
#endif
}

// ---------------------------------------------------------------------------
// fp32 -> bf16 conversion, 4 elements per lane (b128 load, b64 store)
// ---------------------------------------------------------------------------
__global__ __launch_bounds__(256) void cvt_bf16(const float4* __restrict__ in,
                                                v4bf* __restrict__ out, int n4) {
    int i = blockIdx.x * blockDim.x + threadIdx.x;
    int stride = gridDim.x * blockDim.x;
    for (; i < n4; i += stride) {
        float4 v = in[i];
        v4bf o;
        o.x = (__bf16)v.x; o.y = (__bf16)v.y; o.z = (__bf16)v.z; o.w = (__bf16)v.w;
        out[i] = o;
    }
}

// ---------------------------------------------------------------------------
// Product-key retrieval: one block (128 threads = 4 waves) per token,
// one wave32 per head.
// ---------------------------------------------------------------------------
__global__ __launch_bounds__(128) void retrieval_kernel(
    const float* __restrict__ x,      // [NTOK, HID]
    const float* __restrict__ Wq,     // [HEADS*RDIM, HID]
    const float* __restrict__ keys,   // [HEADS, NKEYS, 2, RDIM/2]
    int*   __restrict__ idxb,         // [NTOK, HEADS*TOPK]
    float* __restrict__ pb)           // [NTOK, HEADS*TOPK]
{
    __shared__ float xs[HID];
    __shared__ float qh[HEADS * RDIM];
    __shared__ float sims[HEADS * 2 * NKEYS];

    const int tok  = blockIdx.x;
    const int tid  = threadIdx.x;
    const int h    = tid >> 5;
    const int lane = tid & 31;

    {
        const float4* xr = (const float4*)(x + (size_t)tok * HID);
        float4* xd = (float4*)xs;
        for (int i = tid; i < HID / 4; i += 128) xd[i] = xr[i];
    }
    __syncthreads();

    // q projection: wave h computes its 128 q values (p in {0,1}, n in 0..63)
    for (int i = lane; i < RDIM; i += 32) {
        int p = i >> 6, n = i & 63;
        int row = p * (HEADS * 64) + h * 64 + n;   // reshape (2, HEADS, 64)
        const float* wr = Wq + (size_t)row * HID;
        float acc = 0.f;
        for (int d = 0; d < HID; ++d) acc += xs[d] * wr[d];
        qh[h * RDIM + i] = acc;
    }
    // same-wave LDS producer/consumer: in-order, no barrier needed
    for (int s = lane; s < 2 * NKEYS; s += 32) {
        int p = s >> 7, k = s & 127;
        const float* kv = keys + (((size_t)(h * NKEYS + k) * 2 + p) * (RDIM / 2));
        const float* qv = qh + h * RDIM + p * 64;
        float acc = 0.f;
        for (int n = 0; n < 64; ++n) acc += qv[n] * kv[n];
        sims[h * 2 * NKEYS + s] = acc;
    }

    if (lane == 0) {
        float* sp = sims + h * 2 * NKEYS;
        float sx[TOPK], sy[TOPK];
        int   ix[TOPK], iy[TOPK];
        for (int j = 0; j < TOPK; ++j) {
            float best = -1e30f; int bi = 0;
            for (int k = 0; k < NKEYS; ++k) { float v = sp[k]; if (v > best) { best = v; bi = k; } }
            sx[j] = best; ix[j] = bi; sp[bi] = -1e30f;
        }
        for (int j = 0; j < TOPK; ++j) {
            float best = -1e30f; int bi = 0;
            for (int k = 0; k < NKEYS; ++k) { float v = sp[NKEYS + k]; if (v > best) { best = v; bi = k; } }
            sy[j] = best; iy[j] = bi; sp[NKEYS + bi] = -1e30f;
        }
        float cs[TOPK * TOPK]; int ci[TOPK * TOPK];
        for (int i = 0; i < TOPK; ++i)
            for (int j = 0; j < TOPK; ++j) {
                cs[i * TOPK + j] = sx[i] + sy[j];
                ci[i * TOPK + j] = ix[i] * NKEYS + iy[j];
            }
        float fs[TOPK]; int fi[TOPK];
        for (int j = 0; j < TOPK; ++j) {
            float best = -1e30f; int bi = 0;
            for (int k = 0; k < TOPK * TOPK; ++k) { if (cs[k] > best) { best = cs[k]; bi = k; } }
            fs[j] = best; fi[j] = ci[bi]; cs[bi] = -1e30f;
        }
        float mx = fs[0], sum = 0.f;
        float ex[TOPK];
        for (int j = 0; j < TOPK; ++j) { ex[j] = __expf(fs[j] - mx); sum += ex[j]; }
        float inv = 1.f / sum;
        int base = tok * HEADS * TOPK + h * TOPK;
        for (int j = 0; j < TOPK; ++j) { idxb[base + j] = fi[j]; pb[base + j] = ex[j] * inv; }
    }
}

// ---------------------------------------------------------------------------
// Expert gather+combine: one 256-thread block per token. Initializes d_out.
// ---------------------------------------------------------------------------
__global__ __launch_bounds__(256) void expert_kernel(
    const float* __restrict__ x,
    const float* __restrict__ down,   // [NUM_EXPERTS, HID]
    const float* __restrict__ up,     // [NUM_EXPERTS, HID]
    const int*   __restrict__ idxb,
    const float* __restrict__ pb,
    float* __restrict__ out)
{
    __shared__ float xs[HID];
    __shared__ float dots[32];
    __shared__ float wsc[32];
    __shared__ int   eidx[32];

    const int tok = blockIdx.x;
    const int tid = threadIdx.x;
    {
        const float4* xr = (const float4*)(x + (size_t)tok * HID);
        float4* xd = (float4*)xs;
        if (tid < HID / 4) xd[tid] = xr[tid];
    }
    if (tid < 32) eidx[tid] = idxb[tok * 32 + tid];
    __syncthreads();

    const int wave = tid >> 5, lane = tid & 31;
    for (int e = wave * 4; e < wave * 4 + 4; ++e) {
        const float* dv = down + (size_t)eidx[e] * HID;
        float s = 0.f;
        for (int d = lane; d < HID; d += 32) s += xs[d] * dv[d];
        for (int off = 16; off > 0; off >>= 1) s += __shfl_down(s, off, 32); // wave32
        if (lane == 0) dots[e] = s;
    }
    __syncthreads();
    if (tid < 32) {
        float g = dots[tid];
        wsc[tid] = (g / (1.f + __expf(-g))) * pb[tok * 32 + tid];
    }
    __syncthreads();
    for (int d = tid; d < HID; d += 256) {
        float acc = 0.f;
        for (int e = 0; e < 32; ++e) acc += wsc[e] * up[(size_t)eidx[e] * HID + d];
        out[(size_t)tok * HID + d] = acc;
    }
}

// ---------------------------------------------------------------------------
// WMMA fragment load (ISA 7.12.2 16-bit A/B layout, wave32).
// Lane l: m/n = l&15, half = l>>4; VGPR word r covers K pair k0(r, half).
// `stride` = bf16 elements per LDS row, `kk` = K sub-tile base (0 or 32).
// ---------------------------------------------------------------------------
__device__ __forceinline__ v16bf load_frag(const __bf16* lds, int rowbase,
                                           int mn, int half, int kk, int stride) {
    v16bf f;
    const __bf16* p = lds + (rowbase + mn) * stride + kk;
#pragma unroll
    for (int r = 0; r < 8; ++r) {
        int k0 = (r < 4) ? (2 * r + 8 * half) : (16 + 2 * (r - 4) + 8 * half);
        f[2 * r]     = p[k0];
        f[2 * r + 1] = p[k0 + 1];
    }
    return f;
}

// ---------------------------------------------------------------------------
// GEMM 1: fused gate+up. Block tile 128 x 64, K-tile 64, double-buffered LDS,
// final-tile peeled so the steady-state loop has an unconditional
// s_wait_asynccnt 0x8. 8 waves (4x2), each wave 32x32 for BOTH G and U ->
// 16 wmma per stage. Epilogue: h = silu(g)*u -> bf16.
// ---------------------------------------------------------------------------
__global__ __launch_bounds__(256) void gemm_gateup(
    const __bf16* __restrict__ X,    // [NTOK, HID] bf16
    const __bf16* __restrict__ Wg,   // [INTER, HID] bf16
    const __bf16* __restrict__ Wu,   // [INTER, HID] bf16
    __bf16* __restrict__ H)          // [NTOK, INTER] bf16
{
    __shared__ unsigned int ldsX[2][128 * 32];   // 2 x 16 KB
    __shared__ unsigned int ldsG[2][64 * 32];    // 2 x 8 KB
    __shared__ unsigned int ldsU[2][64 * 32];    // 2 x 8 KB

    const int tid  = threadIdx.x;
    const int n0   = blockIdx.x * 128;
    const int c0   = blockIdx.y * 64;
    const int wave = tid >> 5, lane = tid & 31;
    const int wr   = wave >> 1, wc = wave & 1;
    const int m    = lane & 15, half = lane >> 4;

    v8f cg[2][2], cu[2][2];
#pragma unroll
    for (int i = 0; i < 2; ++i)
#pragma unroll
        for (int j = 0; j < 2; ++j) { cg[i][j] = (v8f){}; cu[i][j] = (v8f){}; }

    const unsigned int* Xw = (const unsigned int*)X;
    const unsigned int* Gw = (const unsigned int*)Wg;
    const unsigned int* Uw = (const unsigned int*)Wu;

    // chunk = 16 B. X tile: 1024 chunks (8/row); W tiles: 512 chunks each.
    int xrow[4], xcw[4];
#pragma unroll
    for (int j = 0; j < 4; ++j) { int c = tid + 256 * j; xrow[j] = c >> 3; xcw[j] = (c & 7) * 4; }
    int grow[2], gcw[2];
#pragma unroll
    for (int j = 0; j < 2; ++j) { int c = tid + 256 * j; grow[j] = c >> 3; gcw[j] = (c & 7) * 4; }

    // 8 async b128 copies per thread per K-tile
    auto issue_tile = [&](int kb, int buf) {
        const int kw = kb >> 1;
#pragma unroll
        for (int j = 0; j < 4; ++j)
            async_copy_b128(Xw + (size_t)(n0 + xrow[j]) * (HID / 2) + kw + xcw[j],
                            ldsX[buf] + xrow[j] * 32 + xcw[j]);
#pragma unroll
        for (int j = 0; j < 2; ++j) {
            async_copy_b128(Gw + (size_t)(c0 + grow[j]) * (HID / 2) + kw + gcw[j],
                            ldsG[buf] + grow[j] * 32 + gcw[j]);
            async_copy_b128(Uw + (size_t)(c0 + grow[j]) * (HID / 2) + kw + gcw[j],
                            ldsU[buf] + grow[j] * 32 + gcw[j]);
        }
    };

    // 16 wmma compute stage on buffer `buf`
    auto compute_tile = [&](int buf) {
        const __bf16* Ax = (const __bf16*)ldsX[buf];
        const __bf16* Bg = (const __bf16*)ldsG[buf];
        const __bf16* Bu = (const __bf16*)ldsU[buf];
#pragma unroll
        for (int kk = 0; kk < 64; kk += 32) {
            v16bf a[2], bg[2], bu[2];
#pragma unroll
            for (int ti = 0; ti < 2; ++ti) a[ti] = load_frag(Ax, wr * 32 + ti * 16, m, half, kk, 64);
#pragma unroll
            for (int tj = 0; tj < 2; ++tj) {
                bg[tj] = load_frag(Bg, wc * 32 + tj * 16, m, half, kk, 64);
                bu[tj] = load_frag(Bu, wc * 32 + tj * 16, m, half, kk, 64);
            }
#pragma unroll
            for (int ti = 0; ti < 2; ++ti)
#pragma unroll
                for (int tj = 0; tj < 2; ++tj) {
                    cg[ti][tj] = __builtin_amdgcn_wmma_f32_16x16x32_bf16(
                        false, a[ti], false, bg[tj], (short)0, cg[ti][tj], false, false);
                    cu[ti][tj] = __builtin_amdgcn_wmma_f32_16x16x32_bf16(
                        false, a[ti], false, bu[tj], (short)0, cu[ti][tj], false, false);
                }
        }
    };

    constexpr int NT = HID / 64;     // 16 K-tiles
    issue_tile(0, 0);
    // steady state: unconditional issue + s_wait_asynccnt 0x8
    for (int t = 0; t < NT - 1; ++t) {
        issue_tile((t + 1) * 64, (t + 1) & 1);
        wait_async_le8();
        __syncthreads();
        compute_tile(t & 1);
        __syncthreads();    // retire reads before this buffer is refilled
    }
    // peeled last tile: drain
    wait_async_le0();
    __syncthreads();
    compute_tile((NT - 1) & 1);

    // epilogue: h = silu(g) * u  (C layout: row = e + 8*half, col = lane&15)
#pragma unroll
    for (int ti = 0; ti < 2; ++ti)
#pragma unroll
        for (int tj = 0; tj < 2; ++tj)
#pragma unroll
            for (int e = 0; e < 8; ++e) {
                int row = n0 + wr * 32 + ti * 16 + e + 8 * half;
                int col = c0 + wc * 32 + tj * 16 + m;
                float g = cg[ti][tj][e], u = cu[ti][tj][e];
                float hv = (g / (1.f + __expf(-g))) * u;
                H[(size_t)row * INTER + col] = (__bf16)hv;
            }
}

// ---------------------------------------------------------------------------
// GEMM 2: out += h @ Wd^T. Block tile 128x128, K-tile 64, double-buffered,
// final-tile peeled; 8 waves (4x2), wave tile 32x64 -> 16 wmma per stage.
// ---------------------------------------------------------------------------
__global__ __launch_bounds__(256) void gemm_down(
    const __bf16* __restrict__ Hm,   // [NTOK, INTER] bf16
    const __bf16* __restrict__ Wd,   // [HID, INTER] bf16
    float* __restrict__ out)         // [NTOK, HID] fp32, accumulated
{
    __shared__ unsigned int ldsA[2][128 * 32];   // 2 x 16 KB
    __shared__ unsigned int ldsB[2][128 * 32];   // 2 x 16 KB

    const int tid  = threadIdx.x;
    const int n0   = blockIdx.x * 128;
    const int c0   = blockIdx.y * 128;
    const int wave = tid >> 5, lane = tid & 31;
    const int wr   = wave >> 1, wc = wave & 1;
    const int m    = lane & 15, half = lane >> 4;

    v8f acc[2][4];
#pragma unroll
    for (int i = 0; i < 2; ++i)
#pragma unroll
        for (int j = 0; j < 4; ++j) acc[i][j] = (v8f){};

    const unsigned int* Aw = (const unsigned int*)Hm;
    const unsigned int* Bw = (const unsigned int*)Wd;

    int crow[4], ccw[4];
#pragma unroll
    for (int j = 0; j < 4; ++j) { int c = tid + 256 * j; crow[j] = c >> 3; ccw[j] = (c & 7) * 4; }

    // 8 async b128 copies per thread per K-tile
    auto issue_tile = [&](int kb, int buf) {
        const int kw = kb >> 1;
#pragma unroll
        for (int j = 0; j < 4; ++j) {
            async_copy_b128(Aw + (size_t)(n0 + crow[j]) * (INTER / 2) + kw + ccw[j],
                            ldsA[buf] + crow[j] * 32 + ccw[j]);
            async_copy_b128(Bw + (size_t)(c0 + crow[j]) * (INTER / 2) + kw + ccw[j],
                            ldsB[buf] + crow[j] * 32 + ccw[j]);
        }
    };

    auto compute_tile = [&](int buf) {
        const __bf16* Ax = (const __bf16*)ldsA[buf];
        const __bf16* Bx = (const __bf16*)ldsB[buf];
#pragma unroll
        for (int kk = 0; kk < 64; kk += 32) {
            v16bf a[2], b[4];
#pragma unroll
            for (int ti = 0; ti < 2; ++ti) a[ti] = load_frag(Ax, wr * 32 + ti * 16, m, half, kk, 64);
#pragma unroll
            for (int tj = 0; tj < 4; ++tj) b[tj] = load_frag(Bx, wc * 64 + tj * 16, m, half, kk, 64);
#pragma unroll
            for (int ti = 0; ti < 2; ++ti)
#pragma unroll
                for (int tj = 0; tj < 4; ++tj)
                    acc[ti][tj] = __builtin_amdgcn_wmma_f32_16x16x32_bf16(
                        false, a[ti], false, b[tj], (short)0, acc[ti][tj], false, false);
        }
    };

    constexpr int NT = INTER / 64;   // 64 K-tiles
    issue_tile(0, 0);
    for (int t = 0; t < NT - 1; ++t) {
        issue_tile((t + 1) * 64, (t + 1) & 1);
        wait_async_le8();
        __syncthreads();
        compute_tile(t & 1);
        __syncthreads();
    }
    wait_async_le0();
    __syncthreads();
    compute_tile((NT - 1) & 1);

#pragma unroll
    for (int ti = 0; ti < 2; ++ti)
#pragma unroll
        for (int tj = 0; tj < 4; ++tj)
#pragma unroll
            for (int e = 0; e < 8; ++e) {
                int row = n0 + wr * 32 + ti * 16 + e + 8 * half;
                int col = c0 + wc * 64 + tj * 16 + m;
                size_t o = (size_t)row * HID + col;
                out[o] += acc[ti][tj][e];
            }
}

// ---------------------------------------------------------------------------
// Host launcher
// ---------------------------------------------------------------------------
extern "C" void kernel_launch(void* const* d_in, const int* in_sizes, int n_in,
                              void* d_out, int out_size, void* d_ws, size_t ws_size,
                              hipStream_t stream) {
    const float* x    = (const float*)d_in[0];   // [NTOK, HID]
    const float* Wq   = (const float*)d_in[1];   // [512, HID]
    const float* keys = (const float*)d_in[2];   // [4,128,2,64]
    const float* down = (const float*)d_in[3];   // [16384, HID]
    const float* up   = (const float*)d_in[4];   // [16384, HID]
    const float* Wg   = (const float*)d_in[5];   // [INTER, HID]
    const float* Wu   = (const float*)d_in[6];   // [INTER, HID]
    const float* Wd   = (const float*)d_in[7];   // [HID, INTER]
    float* out = (float*)d_out;

    char* ws = (char*)d_ws;
    const size_t MB = 1u << 20;
    __bf16* xb  = (__bf16*)(ws + 0 * MB);    // 8 MB
    __bf16* wgb = (__bf16*)(ws + 8 * MB);    // 8 MB
    __bf16* wub = (__bf16*)(ws + 16 * MB);   // 8 MB
    __bf16* wdb = (__bf16*)(ws + 24 * MB);   // 8 MB
    __bf16* hb  = (__bf16*)(ws + 32 * MB);   // 32 MB
    int*    idxb = (int*)(ws + 64 * MB);     // 512 KB
    float*  pbuf = (float*)(ws + 64 * MB + 512 * 1024);

    // 1) bf16 conversions (vectorized x4)
    cvt_bf16<<<1024, 256, 0, stream>>>((const float4*)x,  (v4bf*)xb,  NTOK * HID / 4);
    cvt_bf16<<<2048, 256, 0, stream>>>((const float4*)Wg, (v4bf*)wgb, INTER * HID / 4);
    cvt_bf16<<<2048, 256, 0, stream>>>((const float4*)Wu, (v4bf*)wub, INTER * HID / 4);
    cvt_bf16<<<2048, 256, 0, stream>>>((const float4*)Wd, (v4bf*)wdb, HID * INTER / 4);

    // 2) product-key retrieval (idx + softmax probs)
    retrieval_kernel<<<NTOK, 128, 0, stream>>>(x, Wq, keys, idxb, pbuf);

    // 3) expert gather/combine -> initializes d_out with expert states
    expert_kernel<<<NTOK, 256, 0, stream>>>(x, down, up, idxb, pbuf, out);

    // 4) fused gate/up WMMA GEMM -> h (bf16)
    gemm_gateup<<<dim3(NTOK / 128, INTER / 64), 256, 0, stream>>>(xb, wgb, wub, hb);

    // 5) down WMMA GEMM, accumulate into d_out
    gemm_down<<<dim3(NTOK / 128, HID / 128), 256, 0, stream>>>(hb, wdb, out);
}
